// BilinearCritic_22986664968233
// MI455X (gfx1250) — matrix operand order repaired and checked by reading, not compiled, not run on hardware
//
#include <hip/hip_runtime.h>
#include <hip/hip_bf16.h>
#include <stdint.h>
#include <stddef.h>

// ---------------------------------------------------------------------------
// BilinearCritic on MI455X (gfx1250, wave32, WMMA)
//   B=8192, D_IN=64, D_HID=256, D_EMB=32
//   scores = relu(x@Wx1+bx1)@ (Wx2@W_bil)  @  (relu(y@Wy1+by1)@Wy2+by2)^T + b
//   (W_bil folded into the x-path layer 2; final GEMM is K=32 -> one WMMA)
// ---------------------------------------------------------------------------

typedef __attribute__((ext_vector_type(16))) _Float16 v16h;
typedef __attribute__((ext_vector_type(8)))  _Float16 v8h;
typedef __attribute__((ext_vector_type(8)))  float    v8f;

static __device__ __forceinline__ v16h cat8(v8h lo, v8h hi) {
  return __builtin_shufflevector(lo, hi, 0,1,2,3,4,5,6,7,8,9,10,11,12,13,14,15);
}

#define WMMA16x16x32(A, B, C)                                                  \
  __builtin_amdgcn_wmma_f32_16x16x32_f16(false, (A), false, (B), (short)0,     \
                                         (C), false, false)

static constexpr int BN   = 8192;
static constexpr int DIN  = 64;
static constexpr int DHID = 256;
static constexpr int DEMB = 32;
static constexpr int HPAD = DHID + 8;   // 264 f16: LDS row pad to spread banks

// ---------------------------------------------------------------------------
// Prep kernels: f32 -> f16 conversions + weight transposes + W_bil folding
// ---------------------------------------------------------------------------
__global__ void k_cvt_xy(const float* __restrict__ x, const float* __restrict__ y,
                         _Float16* __restrict__ x16, _Float16* __restrict__ y16,
                         int n) {
  int i = blockIdx.x * blockDim.x + threadIdx.x;
  if (i < n) {
    x16[i] = (_Float16)x[i];
    y16[i] = (_Float16)y[i];
  }
}

// W1T[n*64 + k] = W1[k*256 + n]   (B-operand layout: column-contiguous K)
__global__ void k_prep_w1(const float* __restrict__ Wx1, const float* __restrict__ Wy1,
                          _Float16* __restrict__ W1xT, _Float16* __restrict__ W1yT) {
  int i = blockIdx.x * blockDim.x + threadIdx.x;   // over 256*64
  if (i < DHID * DIN) {
    int n = i / DIN, k = i % DIN;
    W1xT[i] = (_Float16)Wx1[k * DHID + n];
    W1yT[i] = (_Float16)Wy1[k * DHID + n];
  }
}

// W2xT[n*256 + k] = (Wx2 @ W_bil)[k, n]   ;   W2yT[n*256 + k] = Wy2[k, n]
// b2xeff[n] = (bx2 @ W_bil)[n]
__global__ void k_prep_w2(const float* __restrict__ Wx2, const float* __restrict__ bx2,
                          const float* __restrict__ Wy2, const float* __restrict__ Wbil,
                          _Float16* __restrict__ W2xT, _Float16* __restrict__ W2yT,
                          float* __restrict__ b2xeff) {
  int i = blockIdx.x * blockDim.x + threadIdx.x;   // over 32*256
  if (i < DEMB * DHID) {
    int n = i / DHID, k = i % DHID;
    float acc = 0.f;
#pragma unroll
    for (int j = 0; j < DEMB; ++j) acc += Wx2[k * DEMB + j] * Wbil[j * DEMB + n];
    W2xT[i] = (_Float16)acc;
    W2yT[i] = (_Float16)Wy2[k * DEMB + n];
    if (k == 0) {
      float b = 0.f;
#pragma unroll
      for (int j = 0; j < DEMB; ++j) b += bx2[j] * Wbil[j * DEMB + n];
      b2xeff[n] = b;
    }
  }
}

// ---------------------------------------------------------------------------
// Embedding MLP: one wave per 16-row tile. Tiles [0,512) = x-path -> Z16,
// tiles [512,1024) = y-path -> E16y.  Layer1 via WMMA (K=64), ReLU into LDS,
// layer2 via WMMA (K=256, 2 N-tiles).
// ---------------------------------------------------------------------------
__global__ __launch_bounds__(128) void k_embed(
    const _Float16* __restrict__ x16,  const _Float16* __restrict__ y16,
    const _Float16* __restrict__ W1xT, const _Float16* __restrict__ W1yT,
    const float* __restrict__ bx1,     const float* __restrict__ by1,
    const _Float16* __restrict__ W2xT, const _Float16* __restrict__ W2yT,
    const float* __restrict__ b2x,     const float* __restrict__ by2,
    _Float16* __restrict__ Z16,        _Float16* __restrict__ E16y) {
  __shared__ _Float16 lds[4][16 * HPAD];   // 33792 bytes

  const int wave = threadIdx.x >> 5;
  const int lane = threadIdx.x & 31;
  const int r = lane & 15;     // row (A) or column (B/C) within tile
  const int h = lane >> 4;     // lane half

  const int tile = blockIdx.x * 4 + wave;            // 0..1023
  const bool isY = tile >= (BN / 16);
  const int  t   = isY ? tile - (BN / 16) : tile;
  const int  i0  = t * 16;

  const _Float16* in16 = isY ? y16  : x16;
  const _Float16* W1T  = isY ? W1yT : W1xT;
  const float*    b1   = isY ? by1  : bx1;
  const _Float16* W2T  = isY ? W2yT : W2xT;
  const float*    b2   = isY ? by2  : b2x;
  _Float16*       out  = isY ? E16y : Z16;

  // ---- layer 1: H[16,256] = relu(X[16,64] @ W1[64,256] + b1) ----
  const _Float16* arow = in16 + (size_t)(i0 + r) * DIN;
  v16h A0 = cat8(*(const v8h*)(arow +  0 + h * 16), *(const v8h*)(arow +  8 + h * 16));
  v16h A1 = cat8(*(const v8h*)(arow + 32 + h * 16), *(const v8h*)(arow + 40 + h * 16));

#pragma unroll
  for (int n = 0; n < DHID / 16; ++n) {
    const _Float16* brow = W1T + (size_t)(n * 16 + r) * DIN + h * 16;
    v16h B0 = *(const v16h*)(brow);
    v16h B1 = *(const v16h*)(brow + 32);
    v8f c = {};
    c = WMMA16x16x32(A0, B0, c);
    c = WMMA16x16x32(A1, B1, c);
    const float bias = b1[n * 16 + r];
#pragma unroll
    for (int q = 0; q < 8; ++q) {
      float v = c[q] + bias;
      v = v > 0.f ? v : 0.f;
      lds[wave][(q + 8 * h) * HPAD + n * 16 + r] = (_Float16)v;
    }
  }

  // ---- layer 2: E[16,32] = H[16,256] @ W2[256,32] + b2 ----
  v8f cz0 = {}, cz1 = {};
#pragma unroll
  for (int kk = 0; kk < DHID / 32; ++kk) {
    const _Float16* hrow = &lds[wave][r * HPAD + kk * 32];
    v16h A = cat8(*(const v8h*)(hrow + h * 16), *(const v8h*)(hrow + 8 + h * 16));
    const _Float16* bp = W2T + (size_t)r * DHID + kk * 32 + h * 16;
    v16h Bn0 = *(const v16h*)(bp);
    v16h Bn1 = *(const v16h*)(bp + (size_t)16 * DHID);
    cz0 = WMMA16x16x32(A, Bn0, cz0);
    cz1 = WMMA16x16x32(A, Bn1, cz1);
  }
  const float bias0 = b2[r];
  const float bias1 = b2[16 + r];
#pragma unroll
  for (int q = 0; q < 8; ++q) {
    const int row = i0 + q + 8 * h;
    out[(size_t)row * DEMB + r]      = (_Float16)(cz0[q] + bias0);
    out[(size_t)row * DEMB + 16 + r] = (_Float16)(cz1[q] + bias1);
  }
}

// ---------------------------------------------------------------------------
// scores[8192,8192] = Z16 @ E16y^T + b_bil.  One wave -> 16x64 tile
// (A reused across 4 WMMAs). Output streamed with non-temporal stores so the
// 256MB result doesn't evict Z16/ye16 (1MB, L2-resident) from the 192MB L2.
// ---------------------------------------------------------------------------
__global__ __launch_bounds__(256) void k_scores(
    const _Float16* __restrict__ Z16, const _Float16* __restrict__ E16y,
    const float* __restrict__ bbil, float* __restrict__ out) {
  const int wave = threadIdx.x >> 5;
  const int lane = threadIdx.x & 31;
  const int r = lane & 15;
  const int h = lane >> 4;

  const int i0 = blockIdx.y * 128 + wave * 16;
  const int j0 = blockIdx.x * 64;
  const float bb = bbil[0];

  const _Float16* arow = Z16 + (size_t)(i0 + r) * DEMB;
  v16h A = cat8(*(const v8h*)(arow + h * 16), *(const v8h*)(arow + 8 + h * 16));

#pragma unroll
  for (int jt = 0; jt < 4; ++jt) {
    const int j = j0 + jt * 16;
    v16h B = *(const v16h*)(E16y + (size_t)(j + r) * DEMB + h * 16);
    v8f c = {};
    c = WMMA16x16x32(A, B, c);
#pragma unroll
    for (int q = 0; q < 8; ++q) {
      __builtin_nontemporal_store(c[q] + bb,
          &out[(size_t)(i0 + q + 8 * h) * BN + (size_t)(j + r)]);
    }
  }

  if (blockIdx.x == 0 && blockIdx.y == 0 && threadIdx.x == 0)
    out[(size_t)BN * BN] = 0.0f;   // kl_loss
}

// ---------------------------------------------------------------------------
// Host-side launch
// ---------------------------------------------------------------------------
extern "C" void kernel_launch(void* const* d_in, const int* in_sizes, int n_in,
                              void* d_out, int out_size, void* d_ws, size_t ws_size,
                              hipStream_t stream) {
  (void)in_sizes; (void)n_in; (void)out_size; (void)ws_size;

  const float* x    = (const float*)d_in[0];
  const float* y    = (const float*)d_in[1];
  const float* Wx1  = (const float*)d_in[2];
  const float* bx1  = (const float*)d_in[3];
  const float* Wx2  = (const float*)d_in[4];
  const float* bx2  = (const float*)d_in[5];
  const float* Wy1  = (const float*)d_in[6];
  const float* by1  = (const float*)d_in[7];
  const float* Wy2  = (const float*)d_in[8];
  const float* by2  = (const float*)d_in[9];
  const float* Wbil = (const float*)d_in[10];
  const float* bbil = (const float*)d_in[11];
  float* out = (float*)d_out;

  // workspace layout (all chunks >= 32B aligned)
  char* ws = (char*)d_ws;
  size_t o = 0;
  _Float16* x16   = (_Float16*)(ws + o); o += (size_t)BN * DIN * 2;      // 1 MB
  _Float16* y16   = (_Float16*)(ws + o); o += (size_t)BN * DIN * 2;      // 1 MB
  _Float16* W1xT  = (_Float16*)(ws + o); o += (size_t)DHID * DIN * 2;
  _Float16* W1yT  = (_Float16*)(ws + o); o += (size_t)DHID * DIN * 2;
  _Float16* W2xT  = (_Float16*)(ws + o); o += (size_t)DEMB * DHID * 2;
  _Float16* W2yT  = (_Float16*)(ws + o); o += (size_t)DEMB * DHID * 2;
  float*    b2xe  = (float*)   (ws + o); o += 128;
  _Float16* Z16   = (_Float16*)(ws + o); o += (size_t)BN * DEMB * 2;     // 512 KB
  _Float16* E16y  = (_Float16*)(ws + o); o += (size_t)BN * DEMB * 2;     // 512 KB

  // prep
  {
    const int n = BN * DIN;
    k_cvt_xy<<<(n + 255) / 256, 256, 0, stream>>>(x, y, x16, y16, n);
    k_prep_w1<<<(DHID * DIN + 255) / 256, 256, 0, stream>>>(Wx1, Wy1, W1xT, W1yT);
    k_prep_w2<<<(DEMB * DHID + 255) / 256, 256, 0, stream>>>(Wx2, bx2, Wy2, Wbil,
                                                             W2xT, W2yT, b2xe);
  }

  // embeddings: 1024 16-row tiles (512 x-path + 512 y-path), 4 waves/block
  k_embed<<<(2 * BN / 16) / 4, 128, 0, stream>>>(
      x16, y16, W1xT, W1yT, bx1, by1, W2xT, W2yT, b2xe, by2, Z16, E16y);

  // scores: block = 8 waves = 128 rows x 64 cols
  dim3 grid(BN / 64, BN / 128);
  k_scores<<<grid, 256, 0, stream>>>(Z16, E16y, bbil, out);
}
